// Net_17179869915
// MI455X (gfx1250) — 8-bit matrix operand order repaired and checked by reading, compiled, not run on hardware
//
#include <hip/hip_runtime.h>

typedef float v2f __attribute__((ext_vector_type(2)));
typedef float v8f __attribute__((ext_vector_type(8)));
typedef int   v8i __attribute__((ext_vector_type(8)));

#define BATCH  32768
#define NIN    784
#define NH     64
#define NOUT   10
#define RPB    128
#define NBLK   (BATCH / RPB)
#define CHUNKS 25

__device__ __forceinline__ unsigned lds_off(const void* p) {
  return (unsigned)(unsigned long long)p;
}

__global__ __launch_bounds__(256) void k_fc1(const float* __restrict__ x,
                                             const float* __restrict__ w1,
                                             const float* __restrict__ b1,
                                             float* __restrict__ h1,
                                             float* __restrict__ part) {
  __shared__ __align__(16) float Xl[2][RPB * 36];
  __shared__ float Wl[2][32 * 65];
  __shared__ float rsum[8 * 64];
  __shared__ float rsq[8 * 64];

  const int t     = threadIdx.x;
  const int lane  = t & 31;
  const int wave  = t >> 5;
  const int grp   = lane >> 4;
  const int laneM = lane & 15;
  const int rowBlock = blockIdx.x * RPB;
  const int waveRow  = wave * 16;

  v8f acc[4];
  for (int n = 0; n < 4; ++n)
    for (int r = 0; r < 8; ++r) acc[n][r] = 0.0f;

  auto issueX = [&](int c, int p) {
#pragma unroll
    for (int i = 0; i < 4; ++i) {
      int idx = t + i * 256;
      int rr = idx >> 3, c4 = idx & 7;
      int kcol = c * 32 + c4 * 4;
      if (kcol < NIN) {
        unsigned goff = (unsigned)(((unsigned)(rowBlock + rr) * NIN + kcol) * 4u);
        unsigned loff = lds_off(&Xl[p][rr * 36 + c4 * 4]);
        asm volatile("global_load_async_to_lds_b128 %0, %1, %2"
                     :: "v"(loff), "v"(goff), "s"(x) : "memory");
      }
    }
  };
  auto loadW = [&](int c, float* wr) {
#pragma unroll
    for (int i = 0; i < 8; ++i) {
      int idx = t + i * 256;
      int cc = idx & 31, n = idx >> 5;
      int k = c * 32 + cc;
      float s = 0.f;
      if (k < NIN) {
        float v = w1[n * NIN + k];
        s = (v > 0.f) ? 1.f : ((v < 0.f) ? -1.f : 0.f);
      }
      wr[i] = s;
    }
  };

  float wr[8];
  issueX(0, 0);
  loadW(0, wr);

  for (int c = 0; c < CHUNKS; ++c) {
    const int p = c & 1;
#pragma unroll
    for (int i = 0; i < 8; ++i) {
      int idx = t + i * 256;
      int cc = idx & 31, n = idx >> 5;
      Wl[p][cc * 65 + n] = wr[i];
    }
    asm volatile("s_wait_asynccnt 0" ::: "memory");
    __syncthreads();
    if (c + 1 < CHUNKS) {
      issueX(c + 1, p ^ 1);
      loadW(c + 1, wr);
    }
#pragma unroll
    for (int kk = 0; kk < 32; kk += 4) {
      const float* xr = &Xl[p][(waveRow + laneM) * 36 + kk + 2 * grp];
      v2f a;
      a.x = xr[0];
      a.y = xr[1];
#pragma unroll
      for (int n = 0; n < 4; ++n) {
        v2f b;
        b.x = Wl[p][(kk + 2 * grp) * 65 + n * 16 + laneM];
        b.y = Wl[p][(kk + 2 * grp + 1) * 65 + n * 16 + laneM];
        acc[n] = __builtin_amdgcn_wmma_f32_16x16x4_f32(
            false, a, false, b, (short)0, acc[n], false, false);
      }
    }
  }

#pragma unroll
  for (int n = 0; n < 4; ++n) {
    int col = n * 16 + laneM;
    float bias = b1[col];
    float s = 0.f, q = 0.f;
#pragma unroll
    for (int r = 0; r < 8; ++r) {
      float v = acc[n][r] + bias;
      int row = rowBlock + waveRow + r + 8 * grp;
      h1[(size_t)row * NH + col] = v;
      s += v;
      q += v * v;
    }
    s += __shfl_xor(s, 16);
    q += __shfl_xor(q, 16);
    if (grp == 0) { rsum[wave * 64 + col] = s; rsq[wave * 64 + col] = q; }
  }
  __syncthreads();
  if (t < 128) {
    int col = t & 63;
    float a2 = 0.f;
    if (t < 64) {
      for (int w = 0; w < 8; ++w) a2 += rsum[w * 64 + col];
      part[blockIdx.x * 128 + col] = a2;
    } else {
      for (int w = 0; w < 8; ++w) a2 += rsq[w * 64 + col];
      part[blockIdx.x * 128 + 64 + col] = a2;
    }
  }
}

__global__ void k_stats(const float* __restrict__ part,
                        float* __restrict__ stats, int nblocks) {
  int col = threadIdx.x;
  if (col >= 64) return;
  float s = 0.f, q = 0.f;
  for (int b = 0; b < nblocks; ++b) {
    s += part[b * 128 + col];
    q += part[b * 128 + 64 + col];
  }
  const float invB = 1.0f / (float)BATCH;
  float mean = s * invB;
  float var = q * invB - mean * mean;
  stats[col] = mean;
  stats[64 + col] = 1.0f / sqrtf(var + 1e-5f);
}

__global__ __launch_bounds__(256) void k_fc2(const float* __restrict__ h1,
                                             const float* __restrict__ w2,
                                             const float* __restrict__ b2,
                                             const float* __restrict__ stats1,
                                             float* __restrict__ h2,
                                             float* __restrict__ part) {
  __shared__ signed char S[64 * 68];
  __shared__ float meanL[64], rstdL[64];
  __shared__ float rsum[8 * 64], rsq[8 * 64];

  const int t = threadIdx.x, lane = t & 31, wave = t >> 5;
  const int grp = lane >> 4, laneM = lane & 15;

  if (t < 64) { meanL[t] = stats1[t]; rstdL[t] = stats1[64 + t]; }
#pragma unroll
  for (int i = 0; i < 16; ++i) {
    int idx = t + i * 256;
    int k = idx & 63, n = idx >> 6;
    float v = w2[n * 64 + k];
    S[n * 68 + k] = (signed char)((v > 0.f) ? 1 : ((v < 0.f) ? -1 : 0));
  }
  __syncthreads();

  v8i bf[4];
#pragma unroll
  for (int n = 0; n < 4; ++n) {
    int col = n * 16 + laneM;
#pragma unroll
    for (int v = 0; v < 8; ++v) {
      int kb = ((v >= 4) ? 32 : 0) + grp * 16 + (v & 3) * 4;
      bf[n][v] = *(const int*)&S[col * 68 + kb];
    }
  }

  const int row = blockIdx.x * RPB + wave * 16 + laneM;
  v8i af;
#pragma unroll
  for (int v = 0; v < 8; ++v) {
    int kb = ((v >= 4) ? 32 : 0) + grp * 16 + (v & 3) * 4;
    float4 hv = *(const float4*)(h1 + (size_t)row * NH + kb);
    float e[4] = {hv.x, hv.y, hv.z, hv.w};
    int pk = 0;
#pragma unroll
    for (int j = 0; j < 4; ++j) {
      float hn = (e[j] - meanL[kb + j]) * rstdL[kb + j];
      hn = fminf(1.f, fmaxf(-1.f, hn));
      int sb = (hn > 0.f) ? 1 : ((hn < 0.f) ? -1 : 0);
      pk |= (sb & 0xFF) << (8 * j);
    }
    af[v] = pk;
  }

  v8i zero = {0, 0, 0, 0, 0, 0, 0, 0};
  v8i ci[4];
#pragma unroll
  for (int n = 0; n < 4; ++n)
    ci[n] = __builtin_amdgcn_wmma_i32_16x16x64_iu8(true, af, true, bf[n], zero,
                                                   false, false);

#pragma unroll
  for (int n = 0; n < 4; ++n) {
    int col = n * 16 + laneM;
    float bias = b2[col];
    float s = 0.f, q = 0.f;
#pragma unroll
    for (int r = 0; r < 8; ++r) {
      float v = (float)ci[n][r] + bias;
      int orow = blockIdx.x * RPB + wave * 16 + r + 8 * grp;
      h2[(size_t)orow * NH + col] = v;
      s += v;
      q += v * v;
    }
    s += __shfl_xor(s, 16);
    q += __shfl_xor(q, 16);
    if (grp == 0) { rsum[wave * 64 + col] = s; rsq[wave * 64 + col] = q; }
  }
  __syncthreads();
  if (t < 128) {
    int col = t & 63;
    float a2 = 0.f;
    if (t < 64) {
      for (int w = 0; w < 8; ++w) a2 += rsum[w * 64 + col];
      part[blockIdx.x * 128 + col] = a2;
    } else {
      for (int w = 0; w < 8; ++w) a2 += rsq[w * 64 + col];
      part[blockIdx.x * 128 + 64 + col] = a2;
    }
  }
}

__global__ __launch_bounds__(256) void k_fc3(const float* __restrict__ h2,
                                             const float* __restrict__ w3,
                                             const float* __restrict__ b3,
                                             const float* __restrict__ stats2,
                                             float* __restrict__ out) {
  __shared__ float Wl[NOUT * 64];
  __shared__ float bl[NOUT];
  __shared__ float meanL[64], rstdL[64];
  const int t = threadIdx.x;
  for (int i = t; i < NOUT * 64; i += 256) Wl[i] = w3[i];
  if (t < NOUT) bl[t] = b3[t];
  if (t < 64) { meanL[t] = stats2[t]; rstdL[t] = stats2[64 + t]; }
  __syncthreads();

  const int row = blockIdx.x * 256 + t;
  float a[NOUT];
#pragma unroll
  for (int j = 0; j < NOUT; ++j) a[j] = bl[j];
  for (int k = 0; k < 64; k += 4) {
    float4 hv = *(const float4*)(h2 + (size_t)row * NH + k);
    float e[4] = {hv.x, hv.y, hv.z, hv.w};
#pragma unroll
    for (int jj = 0; jj < 4; ++jj) {
      float hn = fminf(1.f, fmaxf(-1.f, (e[jj] - meanL[k + jj]) * rstdL[k + jj]));
#pragma unroll
      for (int j = 0; j < NOUT; ++j) a[j] += hn * Wl[j * 64 + k + jj];
    }
  }
#pragma unroll
  for (int j = 0; j < NOUT; ++j) out[(size_t)row * NOUT + j] = a[j];
}

extern "C" void kernel_launch(void* const* d_in, const int* in_sizes, int n_in,
                              void* d_out, int out_size, void* d_ws,
                              size_t ws_size, hipStream_t stream) {
  (void)in_sizes; (void)n_in; (void)out_size; (void)ws_size;
  const float* x  = (const float*)d_in[0];
  const float* w1 = (const float*)d_in[1];
  const float* b1 = (const float*)d_in[2];
  const float* w2 = (const float*)d_in[3];
  const float* b2 = (const float*)d_in[4];
  const float* w3 = (const float*)d_in[5];
  const float* b3 = (const float*)d_in[6];
  float* out = (float*)d_out;

  float* ws     = (float*)d_ws;
  float* h1     = ws;
  float* h2     = h1 + (size_t)BATCH * NH;
  float* part1  = h2 + (size_t)BATCH * NH;
  float* part2  = part1 + (size_t)NBLK * 128;
  float* stats1 = part2 + (size_t)NBLK * 128;
  float* stats2 = stats1 + 128;

  k_fc1<<<NBLK, 256, 0, stream>>>(x, w1, b1, h1, part1);
  k_stats<<<1, 64, 0, stream>>>(part1, stats1, NBLK);
  k_fc2<<<NBLK, 256, 0, stream>>>(h1, w2, b2, stats1, h2, part2);
  k_stats<<<1, 64, 0, stream>>>(part2, stats2, NBLK);
  k_fc3<<<BATCH / 256, 256, 0, stream>>>(h2, w3, b3, stats2, out);
}
